// MMoEModel_51719996178602
// MI455X (gfx1250) — matrix operand order repaired and checked by reading, not compiled, not run on hardware
//
#include <hip/hip_runtime.h>
#include <hip/hip_bf16.h>

// ---------------------------------------------------------------------------
// MMoE forward for MI455X (gfx1250, wave32, WMMA + async-LDS tile staging).
// Pipeline:
//   0) f32 -> f16 weight conversion into workspace (L2-resident afterwards)
//   1) build f16 emb (B x 1728): embedding gather + numerical (13->64) GEMM
//   2) batched WMMA GEMM (+bias, +ReLU): experts L1/L2, towers L1/L2
//      - 128x64 block tile, 32x32 wave tile (4 accumulators),
//        double-buffered LDS staging via global_load_async_to_lds_b128
//        (ASYNCcnt), ds_load fragment reads
//   3) gate softmax (T x B x 8), gate-weighted expert mixture
//   4) final 64-dot + sigmoid
// ---------------------------------------------------------------------------

typedef __attribute__((ext_vector_type(16))) _Float16 half16;
typedef __attribute__((ext_vector_type(8)))  _Float16 half8;
typedef __attribute__((ext_vector_type(8)))  float    float8;

#define KB   4096   // batch
#define KF   26
#define KED  64
#define KD   1728   // (F+1)*ED
#define KNUM 13
#define KE   8
#define KT   2
#define KB0  512
#define KB1  256
#define KT0  128
#define KT1  64

// ---------------------------------------------------------------- converts
__global__ __launch_bounds__(256) void f32_to_f16_kernel(
    const float* __restrict__ in, _Float16* __restrict__ out, int n) {
  int i = blockIdx.x * 256 + threadIdx.x;
  if (i < n) out[i] = (_Float16)in[i];
}

// ------------------------------------------------------------- build emb
// grid = B blocks, 64 threads; thread t owns column t of each 64-wide slice.
__global__ __launch_bounds__(64) void build_emb_kernel(
    const int* __restrict__ cat, const float* __restrict__ numx,
    const int* __restrict__ offs, const float* __restrict__ W_emb,
    const float* __restrict__ Wn, const float* __restrict__ bn,
    _Float16* __restrict__ emb) {
  int b = blockIdx.x;
  int t = threadIdx.x;  // 0..63
  _Float16* row = emb + (size_t)b * KD;
  const int* cb = cat + (size_t)b * KF;
#pragma unroll 2
  for (int f = 0; f < KF; ++f) {
    int idx = cb[f] + offs[f];
    row[f * KED + t] = (_Float16)W_emb[(size_t)idx * KED + t];
  }
  float acc = bn[t];
  const float* nx = numx + (size_t)b * KNUM;
#pragma unroll
  for (int j = 0; j < KNUM; ++j) acc += nx[j] * Wn[j * KED + t];
  row[KF * KED + t] = (_Float16)acc;
}

__device__ __forceinline__ float8 wmma_f16(half16 a, half16 b, float8 c) {
  return __builtin_amdgcn_wmma_f32_16x16x32_f16(false, a, false, b, (short)0,
                                                c, false, false);
}

// -------------------------------------------------- batched WMMA GEMM
// C[e] = act(A[e] (MxK) * B[e] (KxN) + bias[e]), f16 in, f32 accum, f16 out.
// Block = 256 threads = 8 waves; block tile 128(M) x 64(N), K step 32.
// Wave w: wm = w&3 -> 32-row strip, wn = w>>2 -> 32-col strip; 4 accumulators.
// Double-buffered LDS staging with async copies (3 async ops/wave/tile).
__global__ __launch_bounds__(256) void gemm_bias_relu_wmma(
    const _Float16* __restrict__ A, int lda, long strideA,
    const _Float16* __restrict__ Bm, int ldb, long strideB,
    const float* __restrict__ bias, int strideBias,
    _Float16* __restrict__ C, int ldc, long strideC,
    int K, int relu) {
  __shared__ __attribute__((aligned(64))) _Float16 sA[2][128 * 32];
  __shared__ __attribute__((aligned(64))) _Float16 sB[2][32 * 64];

  const int e = blockIdx.z;
  A    += (size_t)e * strideA;
  Bm   += (size_t)e * strideB;
  bias += (size_t)e * strideBias;
  C    += (size_t)e * strideC;

  const int tid  = threadIdx.x;
  const int lane = tid & 31;
  const int wave = tid >> 5;  // 0..7
  const int wm = wave & 3;    // 0..3 : 32-row strip
  const int wn = wave >> 2;   // 0..1 : 32-col strip
  const int mblk = blockIdx.x * 128;
  const int nblk = blockIdx.y * 64;

  // Per-thread 16B chunks of each staged tile.
  // A tile: 128 rows x 4 chunks = 512 chunks -> 2 per thread (q, q+256).
  // B tile: 32 rows x 8 chunks = 256 chunks -> 1 per thread.
  const int a0r = tid >> 2, a0c = (tid & 3) << 3;
  const int a1r = a0r + 64;
  const int br = tid >> 3, bc = (tid & 7) << 3;
  const _Float16* gA0 = A + (size_t)(mblk + a0r) * lda + a0c;
  const _Float16* gA1 = A + (size_t)(mblk + a1r) * lda + a0c;
  const _Float16* gB  = Bm + (size_t)br * ldb + nblk + bc;
  const unsigned ldsA0[2] = {(unsigned)(size_t)&sA[0][a0r * 32 + a0c],
                             (unsigned)(size_t)&sA[1][a0r * 32 + a0c]};
  const unsigned ldsA1[2] = {(unsigned)(size_t)&sA[0][a1r * 32 + a0c],
                             (unsigned)(size_t)&sA[1][a1r * 32 + a0c]};
  const unsigned ldsB[2]  = {(unsigned)(size_t)&sB[0][br * 64 + bc],
                             (unsigned)(size_t)&sB[1][br * 64 + bc]};

  // LDS fragment pointers (per wave/lane). A rows wm*32 + (lane&15) (+16).
  const int afr = (wm * 32 + (lane & 15)) * 32 + ((lane >> 4) << 3);
  const _Float16* fA0p[2] = {&sA[0][afr], &sA[1][afr]};
  const _Float16* fA1p[2] = {&sA[0][afr + 16 * 32], &sA[1][afr + 16 * 32]};
  const _Float16* fBp[2]  = {&sB[0][((lane >> 4) << 3) * 64 + wn * 32 + (lane & 15)],
                             &sB[1][((lane >> 4) << 3) * 64 + wn * 32 + (lane & 15)]};

  const int nk = K >> 5;

  // Issue async tile copy (3 async ops per wave per tile).
  auto issue_tile = [&](int kt, int p) {
    const _Float16* a0 = gA0 + (size_t)kt * 32;
    const _Float16* a1 = gA1 + (size_t)kt * 32;
    const _Float16* b  = gB + (size_t)kt * 32 * ldb;
    asm volatile("global_load_async_to_lds_b128 %0, %1, off"
                 :: "v"(ldsA0[p]), "v"(a0) : "memory");
    asm volatile("global_load_async_to_lds_b128 %0, %1, off"
                 :: "v"(ldsA1[p]), "v"(a1) : "memory");
    asm volatile("global_load_async_to_lds_b128 %0, %1, off"
                 :: "v"(ldsB[p]), "v"(b) : "memory");
  };

  issue_tile(0, 0);

  float8 c00 = {}, c01 = {}, c10 = {}, c11 = {};

  for (int kt = 0; kt < nk; ++kt) {
    const int p = kt & 1;
    if (kt + 1 < nk) {
      issue_tile(kt + 1, p ^ 1);  // overlap copy of next tile with compute
      // 3 newest ops (next tile) may remain outstanding; tile kt is done.
      asm volatile("s_wait_asynccnt 0x3" ::: "memory");
    } else {
      asm volatile("s_wait_asynccnt 0x0" ::: "memory");
    }
    __syncthreads();  // all waves' portions of tile kt visible

    union { half16 v; half8 h[2]; } a0, a1;
    a0.h[0] = *(const half8*)(fA0p[p]);
    a0.h[1] = *(const half8*)(fA0p[p] + 16);
    a1.h[0] = *(const half8*)(fA1p[p]);
    a1.h[1] = *(const half8*)(fA1p[p] + 16);
    half16 b0, b1;
#pragma unroll
    for (int s = 0; s < 16; ++s) {
      const int ko = (s < 8 ? s : s + 8) * 64;
      b0[s] = fBp[p][ko];
      b1[s] = fBp[p][ko + 16];
    }

    c00 = wmma_f16(a0.v, b0, c00);
    c01 = wmma_f16(a0.v, b1, c01);
    c10 = wmma_f16(a1.v, b0, c10);
    c11 = wmma_f16(a1.v, b1, c11);

    __syncthreads();  // protect buffer p before it is overwritten (kt+2)
  }

  // Epilogue: acc cXY element r -> row mblk + wm*32 + X*16 + r + 8*(lane>>4),
  //           col nblk + wn*32 + Y*16 + (lane&15)
  const int n0 = nblk + wn * 32;
  const int cn = n0 + (lane & 15);
  const int m0 = mblk + wm * 32 + ((lane >> 4) << 3);
  const float bi0 = bias[cn];
  const float bi1 = bias[cn + 16];
#pragma unroll
  for (int r = 0; r < 8; ++r) {
    float v00 = c00[r] + bi0;
    float v01 = c01[r] + bi1;
    float v10 = c10[r] + bi0;
    float v11 = c11[r] + bi1;
    if (relu) {
      v00 = fmaxf(v00, 0.f);
      v01 = fmaxf(v01, 0.f);
      v10 = fmaxf(v10, 0.f);
      v11 = fmaxf(v11, 0.f);
    }
    size_t row0 = (size_t)(m0 + r) * ldc;
    size_t row1 = (size_t)(m0 + 16 + r) * ldc;
    C[row0 + cn]      = (_Float16)v00;
    C[row0 + cn + 16] = (_Float16)v01;
    C[row1 + cn]      = (_Float16)v10;
    C[row1 + cn + 16] = (_Float16)v11;
  }
}

// ------------------------------------------------------------ gate softmax
// one thread per (t,b): 8 dots of length D, softmax over E=8.
__global__ __launch_bounds__(256) void gate_softmax_kernel(
    const _Float16* __restrict__ emb, const float* __restrict__ Wg,
    const float* __restrict__ bg, float* __restrict__ gate) {
  int i = blockIdx.x * 256 + threadIdx.x;
  if (i >= KT * KB) return;
  int t = i / KB, b = i % KB;
  const _Float16* er = emb + (size_t)b * KD;
  const float* W = Wg + (size_t)t * KD * KE;
  float acc[KE];
#pragma unroll
  for (int e = 0; e < KE; ++e) acc[e] = bg[t * KE + e];
  for (int d = 0; d < KD; ++d) {
    float ev = (float)er[d];
#pragma unroll
    for (int e = 0; e < KE; ++e) acc[e] += ev * W[(size_t)d * KE + e];
  }
  float mx = acc[0];
#pragma unroll
  for (int e = 1; e < KE; ++e) mx = fmaxf(mx, acc[e]);
  float s = 0.f;
#pragma unroll
  for (int e = 0; e < KE; ++e) { acc[e] = __expf(acc[e] - mx); s += acc[e]; }
  float inv = 1.f / s;
  float* g = gate + ((size_t)t * KB + b) * KE;
#pragma unroll
  for (int e = 0; e < KE; ++e) g[e] = acc[e] * inv;
}

// ------------------------------------------- gate-weighted expert mixture
// task_fea[t,b,o] = sum_e gate[t,b,e] * fea[e,b,o]
__global__ __launch_bounds__(256) void task_fea_kernel(
    const float* __restrict__ gate, const _Float16* __restrict__ fea,
    _Float16* __restrict__ tf) {
  int i = blockIdx.x * 256 + threadIdx.x;
  if (i >= KT * KB * KB1) return;
  int t = i / (KB * KB1);
  int rem = i % (KB * KB1);
  int b = rem / KB1;
  int o = rem % KB1;
  const float* g = gate + ((size_t)t * KB + b) * KE;
  float acc = 0.f;
#pragma unroll
  for (int e = 0; e < KE; ++e)
    acc += g[e] * (float)fea[((size_t)e * KB + b) * KB1 + o];
  tf[i] = (_Float16)acc;
}

// -------------------------------------------------- final dot + sigmoid
__global__ __launch_bounds__(256) void final_kernel(
    const _Float16* __restrict__ h2, const float* __restrict__ Wt3,
    const float* __restrict__ bt3, float* __restrict__ out) {
  int i = blockIdx.x * 256 + threadIdx.x;
  if (i >= KT * KB) return;
  int t = i / KB, b = i % KB;
  const _Float16* r = h2 + ((size_t)t * KB + b) * KT1;
  const float* w = Wt3 + t * KT1;
  float acc = bt3[t];
#pragma unroll 4
  for (int k = 0; k < KT1; ++k) acc += (float)r[k] * w[k];
  out[i] = 1.f / (1.f + __expf(-acc));
}

// ---------------------------------------------------------------------------
extern "C" void kernel_launch(void* const* d_in, const int* in_sizes, int n_in,
                              void* d_out, int out_size, void* d_ws,
                              size_t ws_size, hipStream_t stream) {
  const int*   cat   = (const int*)d_in[0];
  const float* numx  = (const float*)d_in[1];
  const int*   offs  = (const int*)d_in[2];
  const float* W_emb = (const float*)d_in[3];
  const float* Wn    = (const float*)d_in[4];
  const float* bn    = (const float*)d_in[5];
  const float* We1   = (const float*)d_in[6];
  const float* be1   = (const float*)d_in[7];
  const float* We2   = (const float*)d_in[8];
  const float* be2   = (const float*)d_in[9];
  const float* Wg    = (const float*)d_in[10];
  const float* bg    = (const float*)d_in[11];
  const float* Wt1   = (const float*)d_in[12];
  const float* bt1   = (const float*)d_in[13];
  const float* Wt2   = (const float*)d_in[14];
  const float* bt2   = (const float*)d_in[15];
  const float* Wt3   = (const float*)d_in[16];
  const float* bt3   = (const float*)d_in[17];
  float* out = (float*)d_out;

  // workspace carve (256B aligned)
  char* p = (char*)d_ws;
  auto carve = [&](size_t bytes) {
    void* r = (void*)p;
    p += (bytes + 255) & ~(size_t)255;
    return r;
  };
  _Float16* emb  = (_Float16*)carve((size_t)KB * KD * 2);
  _Float16* we1h = (_Float16*)carve((size_t)KE * KD * KB0 * 2);
  _Float16* we2h = (_Float16*)carve((size_t)KE * KB0 * KB1 * 2);
  _Float16* wt1h = (_Float16*)carve((size_t)KT * KB1 * KT0 * 2);
  _Float16* wt2h = (_Float16*)carve((size_t)KT * KT0 * KT1 * 2);
  _Float16* H    = (_Float16*)carve((size_t)KE * KB * KB0 * 2);
  _Float16* FEA  = (_Float16*)carve((size_t)KE * KB * KB1 * 2);
  float*    GATE = (float*)carve((size_t)KT * KB * KE * 4);
  _Float16* TF   = (_Float16*)carve((size_t)KT * KB * KB1 * 2);
  _Float16* H1   = (_Float16*)carve((size_t)KT * KB * KT0 * 2);
  _Float16* H2   = (_Float16*)carve((size_t)KT * KB * KT1 * 2);

  // 0) weight conversion
  {
    int n;
    n = KE * KD * KB0;
    f32_to_f16_kernel<<<(n + 255) / 256, 256, 0, stream>>>(We1, we1h, n);
    n = KE * KB0 * KB1;
    f32_to_f16_kernel<<<(n + 255) / 256, 256, 0, stream>>>(We2, we2h, n);
    n = KT * KB1 * KT0;
    f32_to_f16_kernel<<<(n + 255) / 256, 256, 0, stream>>>(Wt1, wt1h, n);
    n = KT * KT0 * KT1;
    f32_to_f16_kernel<<<(n + 255) / 256, 256, 0, stream>>>(Wt2, wt2h, n);
  }

  // 1) emb (gather + numerical GEMM), f16
  build_emb_kernel<<<KB, 64, 0, stream>>>(cat, numx, offs, W_emb, Wn, bn, emb);

  // 2) experts layer 1: H[e] = relu(emb @ We1[e] + be1[e])   (4096x1728x512)
  gemm_bias_relu_wmma<<<dim3(KB / 128, KB0 / 64, KE), 256, 0, stream>>>(
      emb, KD, 0L, we1h, KB0, (long)KD * KB0, be1, KB0, H, KB0,
      (long)KB * KB0, KD, 1);

  // 3) experts layer 2: FEA[e] = relu(H[e] @ We2[e] + be2[e]) (4096x512x256)
  gemm_bias_relu_wmma<<<dim3(KB / 128, KB1 / 64, KE), 256, 0, stream>>>(
      H, KB0, (long)KB * KB0, we2h, KB1, (long)KB0 * KB1, be2, KB1, FEA, KB1,
      (long)KB * KB1, KB0, 1);

  // 4) gates
  gate_softmax_kernel<<<(KT * KB + 255) / 256, 256, 0, stream>>>(emb, Wg, bg,
                                                                 GATE);
  // 5) gate-weighted mixture
  task_fea_kernel<<<(KT * KB * KB1 + 255) / 256, 256, 0, stream>>>(GATE, FEA,
                                                                   TF);
  // 6) tower layer 1: H1 = relu(TF @ Wt1 + bt1)  (4096x256x128, batched T)
  gemm_bias_relu_wmma<<<dim3(KB / 128, KT0 / 64, KT), 256, 0, stream>>>(
      TF, KB1, (long)KB * KB1, wt1h, KT0, (long)KB1 * KT0, bt1, KT0, H1, KT0,
      (long)KB * KT0, KB1, 1);

  // 7) tower layer 2: H2 = relu(H1 @ Wt2 + bt2)  (4096x128x64, batched T)
  gemm_bias_relu_wmma<<<dim3(KB / 128, KT1 / 64, KT), 256, 0, stream>>>(
      H1, KT0, (long)KB * KT0, wt2h, KT1, (long)KT0 * KT1, bt2, KT1, H2, KT1,
      (long)KB * KT1, KT0, 1);

  // 8) final dot + sigmoid -> out (T x B)
  final_kernel<<<(KT * KB + 255) / 256, 256, 0, stream>>>(H2, Wt3, bt3, out);
}
